// ViTLayer_1365799600555
// MI455X (gfx1250) — compile-verified, matrix operand order checked by reading, not run on hardware
//
#include <hip/hip_runtime.h>

typedef _Float16 half_t;
typedef __attribute__((ext_vector_type(16))) _Float16 v16h;
typedef __attribute__((ext_vector_type(8)))  _Float16 v8h;
typedef __attribute__((ext_vector_type(8)))  float    v8f;
typedef __attribute__((ext_vector_type(4)))  int      v4i;

#if __has_builtin(__builtin_amdgcn_global_load_async_to_lds_b128)
#define HAS_ASYNC_LDS 1
#else
#define HAS_ASYNC_LDS 0
#endif

typedef __attribute__((address_space(1))) v4i gv4i;   // global 128-bit chunk
typedef __attribute__((address_space(3))) v4i lv4i;   // LDS 128-bit chunk

#define B_   4
#define S_   1024
#define D_   768
#define H_   12
#define DH_  64
#define HID_ 3072
#define M_   (B_*S_)
#define EPS_ 1e-5f
#define INV_SQRT_DH_ 0.125f

#define WMMA_F16(A,Bm,C) \
  __builtin_amdgcn_wmma_f32_16x16x32_f16(false,(A),false,(Bm),(short)0,(C),false,false)

#if HAS_ASYNC_LDS
static __device__ inline void async_copy16(const half_t* g, half_t* l) {
  __builtin_amdgcn_global_load_async_to_lds_b128((gv4i*)(void*)(const void*)g,
                                                 (lv4i*)(void*)l, 0, 0);
}
#endif

static __device__ inline void wait_async0() {
#if __has_builtin(__builtin_amdgcn_s_wait_asynccnt)
  __builtin_amdgcn_s_wait_asynccnt(0);
#else
  asm volatile("s_wait_asynccnt 0x0" ::: "memory");
#endif
}

static __device__ inline v8f zero8() {
  v8f z;
#pragma unroll
  for (int i = 0; i < 8; ++i) z[i] = 0.0f;
  return z;
}

static __device__ inline v16h negh(v16h a) {
  v16h r;
#pragma unroll
  for (int i = 0; i < 16; ++i) r[i] = -a[i];
  return r;
}

// A/B fragment for v_wmma_f32_16x16x32_f16, 16x32 row-major source.
// Lane l: m = l&15, half = l>>4; halves h*8..h*8+7 and 16+h*8..16+h*8+7.
static __device__ inline v16h load_ab_frag(const half_t* base, int stride, int lane) {
  int m = lane & 15, h = lane >> 4;
  const half_t* row = base + (size_t)m * stride;
  v8h a = *(const v8h*)(row + h * 8);
  v8h b = *(const v8h*)(row + 16 + h * 8);
  v16h r;
#pragma unroll
  for (int i = 0; i < 8; ++i) { r[i] = a[i]; r[i + 8] = b[i]; }
  return r;
}

// B fragment where B = 32 K-rows x 16 cols, row-major with row stride `stride`
// (used for attn x V: rows = keys, cols = head dims).
static __device__ inline v16h load_b_colfrag(const half_t* base, int stride, int lane) {
  int n = lane & 15, h = lane >> 4;
  v16h r;
#pragma unroll
  for (int j = 0; j < 8; ++j) {
    int kb = (j < 4) ? (h * 8 + 2 * j) : (16 + h * 8 + 2 * (j - 4));
    r[2 * j]     = base[(size_t)kb * stride + n];
    r[2 * j + 1] = base[(size_t)(kb + 1) * stride + n];
  }
  return r;
}

static __device__ inline float wave_sum32(float v) {
#pragma unroll
  for (int m = 1; m < 32; m <<= 1) v += __shfl_xor(v, m, 32);
  return v;
}
static __device__ inline float grp16_max(float v) {
#pragma unroll
  for (int m = 1; m < 16; m <<= 1) v = fmaxf(v, __shfl_xor(v, m, 32));
  return v;
}
static __device__ inline float grp16_sum(float v) {
#pragma unroll
  for (int m = 1; m < 16; m <<= 1) v += __shfl_xor(v, m, 32);
  return v;
}

static __device__ inline float gelu_tanh(float x) {
  float x3 = x * x * x;
  return 0.5f * x * (1.0f + tanhf(0.7978845608028654f * (x + 0.044715f * x3)));
}

// ---------------------------------------------------------------------------
// deinterleave [n,2] f32 -> planar f32
__global__ void deint_kernel(const float* __restrict__ x,
                             float* __restrict__ xr, float* __restrict__ xi, int n) {
  int i = blockIdx.x * 256 + threadIdx.x;
  if (i < n) { xr[i] = x[2 * i]; xi[i] = x[2 * i + 1]; }
}

// interleaved f32 weights -> planar f16
__global__ void wconv_kernel(const float* __restrict__ w,
                             half_t* __restrict__ wr, half_t* __restrict__ wi, int n) {
  int i = blockIdx.x * 256 + threadIdx.x;
  if (i < n) { wr[i] = (half_t)w[2 * i]; wi[i] = (half_t)w[2 * i + 1]; }
}

// ---------------------------------------------------------------------------
// Complex LayerNorm (covariance whitening). One wave per token row.
__global__ __launch_bounds__(256)
void cln_kernel(const float* __restrict__ xr, const float* __restrict__ xi,
                const float* __restrict__ gamma, const float* __restrict__ beta,
                half_t* __restrict__ outr, half_t* __restrict__ outi, int Dn) {
  int wave = threadIdx.x >> 5, lane = threadIdx.x & 31;
  int row = blockIdx.x * 8 + wave;
  const float* pr = xr + (size_t)row * Dn;
  const float* pi = xi + (size_t)row * Dn;

  float sr = 0.f, si = 0.f;
  for (int d = lane; d < Dn; d += 32) { sr += pr[d]; si += pi[d]; }
  sr = wave_sum32(sr); si = wave_sum32(si);
  float invD = 1.0f / (float)Dn;
  float mr = sr * invD, mi = si * invD;

  float vrr = 0.f, vii = 0.f, vri = 0.f;
  for (int d = lane; d < Dn; d += 32) {
    float cr = pr[d] - mr, ci = pi[d] - mi;
    vrr += cr * cr; vii += ci * ci; vri += cr * ci;
  }
  vrr = wave_sum32(vrr) * invD + EPS_;
  vii = wave_sum32(vii) * invD + EPS_;
  vri = wave_sum32(vri) * invD;

  float s = sqrtf(vrr * vii - vri * vri);
  float t = sqrtf(vrr + vii + 2.0f * s);
  float ist = 1.0f / (s * t);
  float wrr = (vii + s) * ist;
  float wii = (vrr + s) * ist;
  float wri = -vri * ist;

  for (int d = lane; d < Dn; d += 32) {
    float cr = pr[d] - mr, ci = pi[d] - mi;
    float nr = wrr * cr + wri * ci;
    float ni = wri * cr + wii * ci;
    const float* g = gamma + (size_t)d * 4;
    float orr = g[0] * nr + g[1] * ni + beta[2 * d];
    float oii = g[2] * nr + g[3] * ni + beta[2 * d + 1];
    outr[(size_t)row * Dn + d] = (half_t)orr;
    outi[(size_t)row * Dn + d] = (half_t)oii;
  }
}

// ---------------------------------------------------------------------------
// Complex GEMM: Y[M,N] = A[M,K] (complex) x W[N,K]^T (complex) + bias, epilogue.
// Tile staging uses GLOBAL_LOAD_ASYNC_TO_LDS_B128 (ASYNCcnt) when available,
// plus global_prefetch of the next k-tile.
// mode 0: store f16 planar   (QKV)
// mode 1: +residual, store f32 planar  (attn O proj)
// mode 2: CGELU, store f32 planar      (FFN1)
// mode 3: CGELU + residual, store interleaved f32 to d_out (FFN2)
__global__ __launch_bounds__(256)
void cgemm_kernel(const half_t* __restrict__ Ar, const half_t* __restrict__ Ai,
                  const half_t* __restrict__ Wr, const half_t* __restrict__ Wi,
                  const float* __restrict__ bias,
                  int M, int N, int K, int mode,
                  const float* __restrict__ resR, const float* __restrict__ resI,
                  half_t* __restrict__ outHr, half_t* __restrict__ outHi,
                  float* __restrict__ outFr, float* __restrict__ outFi,
                  float* __restrict__ outInter) {
  __shared__ half_t As[2][128 * 32];
  __shared__ half_t Bs[2][64 * 32];

  int tid = threadIdx.x;
  int lane = tid & 31, wave = tid >> 5;
  int wm = (wave >> 1) * 32;       // 0,32,64,96
  int wn = (wave & 1) * 32;        // 0,32
  int m0 = blockIdx.y * 128;
  int n0 = blockIdx.x * 64;

  v8f accR[2][2], accI[2][2];
#pragma unroll
  for (int a = 0; a < 2; ++a)
#pragma unroll
    for (int b = 0; b < 2; ++b) { accR[a][b] = zero8(); accI[a][b] = zero8(); }

  int Ksteps = K >> 5;
  for (int kt = 0; kt < Ksteps; ++kt) {
    // stage A: 128x32 halfs/plane = 512 16B chunks
#pragma unroll
    for (int v = tid; v < 512; v += 256) {
      int row = v >> 2, c8 = v & 3;
      size_t g = (size_t)(m0 + row) * K + (size_t)kt * 32 + c8 * 8;
#if HAS_ASYNC_LDS
      async_copy16(&Ar[g], &As[0][row * 32 + c8 * 8]);
      async_copy16(&Ai[g], &As[1][row * 32 + c8 * 8]);
#else
      *(v8h*)&As[0][row * 32 + c8 * 8] = *(const v8h*)&Ar[g];
      *(v8h*)&As[1][row * 32 + c8 * 8] = *(const v8h*)&Ai[g];
#endif
      if (kt + 1 < Ksteps) {
        __builtin_prefetch(&Ar[g + 32], 0, 0);
        __builtin_prefetch(&Ai[g + 32], 0, 0);
      }
    }
    // stage B: 64x32 halfs/plane = 256 16B chunks
    {
      int v = tid;
      int row = v >> 2, c8 = v & 3;
      size_t g = (size_t)(n0 + row) * K + (size_t)kt * 32 + c8 * 8;
#if HAS_ASYNC_LDS
      async_copy16(&Wr[g], &Bs[0][row * 32 + c8 * 8]);
      async_copy16(&Wi[g], &Bs[1][row * 32 + c8 * 8]);
#else
      *(v8h*)&Bs[0][row * 32 + c8 * 8] = *(const v8h*)&Wr[g];
      *(v8h*)&Bs[1][row * 32 + c8 * 8] = *(const v8h*)&Wi[g];
#endif
      if (kt + 1 < Ksteps) {
        __builtin_prefetch(&Wr[g + 32], 0, 0);
        __builtin_prefetch(&Wi[g + 32], 0, 0);
      }
    }
#if HAS_ASYNC_LDS
    wait_async0();
#endif
    __syncthreads();

    v16h aR[2], aI[2], aIn[2], bR[2], bI[2];
#pragma unroll
    for (int mt = 0; mt < 2; ++mt) {
      aR[mt]  = load_ab_frag(&As[0][0] + (wm + mt * 16) * 32, 32, lane);
      aI[mt]  = load_ab_frag(&As[1][0] + (wm + mt * 16) * 32, 32, lane);
      aIn[mt] = negh(aI[mt]);
    }
#pragma unroll
    for (int nt = 0; nt < 2; ++nt) {
      bR[nt] = load_ab_frag(&Bs[0][0] + (wn + nt * 16) * 32, 32, lane);
      bI[nt] = load_ab_frag(&Bs[1][0] + (wn + nt * 16) * 32, 32, lane);
    }
#pragma unroll
    for (int mt = 0; mt < 2; ++mt)
#pragma unroll
      for (int nt = 0; nt < 2; ++nt) {
        accR[mt][nt] = WMMA_F16(aR[mt],  bR[nt], accR[mt][nt]);
        accR[mt][nt] = WMMA_F16(aIn[mt], bI[nt], accR[mt][nt]);  // - Xi*Wi
        accI[mt][nt] = WMMA_F16(aR[mt],  bI[nt], accI[mt][nt]);
        accI[mt][nt] = WMMA_F16(aI[mt],  bR[nt], accI[mt][nt]);
      }
    __syncthreads();
  }

  int lane15 = lane & 15, halfid = lane >> 4;
#pragma unroll
  for (int mt = 0; mt < 2; ++mt)
#pragma unroll
    for (int nt = 0; nt < 2; ++nt)
#pragma unroll
      for (int r = 0; r < 8; ++r) {
        int m = m0 + wm + mt * 16 + r + halfid * 8;
        int n = n0 + wn + nt * 16 + lane15;
        float yr = accR[mt][nt][r] + bias[2 * n];
        float yi = accI[mt][nt][r] + bias[2 * n + 1];
        size_t o = (size_t)m * N + n;
        if (mode == 0) {
          outHr[o] = (half_t)yr; outHi[o] = (half_t)yi;
        } else if (mode == 1) {
          outFr[o] = yr + resR[o]; outFi[o] = yi + resI[o];
        } else if (mode == 2) {
          outFr[o] = gelu_tanh(yr); outFi[o] = gelu_tanh(yi);
        } else {
          float vr_ = gelu_tanh(yr) + resR[o];
          float vi_ = gelu_tanh(yi) + resI[o];
          outInter[2 * o] = vr_; outInter[2 * o + 1] = vi_;
        }
      }
}

// ---------------------------------------------------------------------------
// Complex attention, flash style. One wave handles 16 queries of one (b,h).
// scores = |q . conj(k)| * 1/sqrt(dh); real softmax; out = attn x v (complex).
__global__ __launch_bounds__(32)
void cattn_kernel(const half_t* __restrict__ qr, const half_t* __restrict__ qi,
                  const half_t* __restrict__ kr, const half_t* __restrict__ ki,
                  const half_t* __restrict__ vr, const half_t* __restrict__ vi,
                  half_t* __restrict__ outr, half_t* __restrict__ outi) {
  const int Dm = D_;
  int bh = blockIdx.x;
  int qb = blockIdx.y;
  int b = bh / H_, hh = bh % H_;
  int lane = threadIdx.x;
  int lane15 = lane & 15, halfid = lane >> 4;
  int qbase = b * S_ + qb * 16;
  int coff = hh * DH_;

  __shared__ half_t Plds[16 * 32];

  // Q fragments (16 queries x 64 dh), and negated Qr for the Im accumulation.
  v16h Qr[2], Qi[2], Qrn[2];
#pragma unroll
  for (int c = 0; c < 2; ++c) {
    Qr[c]  = load_ab_frag(qr + (size_t)qbase * Dm + coff + c * 32, Dm, lane);
    Qi[c]  = load_ab_frag(qi + (size_t)qbase * Dm + coff + c * 32, Dm, lane);
    Qrn[c] = negh(Qr[c]);
  }

  float mrow[8], lrow[8];
  v8f oR[4], oI[4];
#pragma unroll
  for (int r = 0; r < 8; ++r) { mrow[r] = -1e30f; lrow[r] = 0.0f; }
#pragma unroll
  for (int nc = 0; nc < 4; ++nc) { oR[nc] = zero8(); oI[nc] = zero8(); }

  for (int kb = 0; kb < S_ / 32; ++kb) {
    int kbase = b * S_ + kb * 32;
    float sv[2][8];

#pragma unroll
    for (int ks = 0; ks < 2; ++ks) {
      v8f aRe = zero8(), aIm = zero8();
#pragma unroll
      for (int c = 0; c < 2; ++c) {
        v16h Kf  = load_ab_frag(kr + (size_t)(kbase + ks * 16) * Dm + coff + c * 32, Dm, lane);
        v16h Kif = load_ab_frag(ki + (size_t)(kbase + ks * 16) * Dm + coff + c * 32, Dm, lane);
        aRe = WMMA_F16(Qr[c],  Kf,  aRe);   //  qr.kr
        aRe = WMMA_F16(Qi[c],  Kif, aRe);   // +qi.ki
        aIm = WMMA_F16(Qi[c],  Kf,  aIm);   //  qi.kr
        aIm = WMMA_F16(Qrn[c], Kif, aIm);   // -qr.ki
      }
#pragma unroll
      for (int r = 0; r < 8; ++r) {
        float re = aRe[r], im = aIm[r];
        sv[ks][r] = sqrtf(re * re + im * im) * INV_SQRT_DH_;
      }
    }

    // online softmax update (rows replicated across 16-lane groups)
    float scl[8], p0[8], p1[8];
#pragma unroll
    for (int r = 0; r < 8; ++r) {
      float mx = grp16_max(fmaxf(sv[0][r], sv[1][r]));
      float mnew = fmaxf(mrow[r], mx);
      scl[r] = __expf(mrow[r] - mnew);
      p0[r] = __expf(sv[0][r] - mnew);
      p1[r] = __expf(sv[1][r] - mnew);
      float ps = grp16_sum(p0[r] + p1[r]);
      lrow[r] = lrow[r] * scl[r] + ps;
      mrow[r] = mnew;
    }
#pragma unroll
    for (int nc = 0; nc < 4; ++nc)
#pragma unroll
      for (int r = 0; r < 8; ++r) { oR[nc][r] *= scl[r]; oI[nc][r] *= scl[r]; }

    // transpose P (C layout -> A layout) through LDS
#pragma unroll
    for (int r = 0; r < 8; ++r) {
      Plds[(r + 8 * halfid) * 32 + 0 * 16 + lane15] = (half_t)p0[r];
      Plds[(r + 8 * halfid) * 32 + 1 * 16 + lane15] = (half_t)p1[r];
    }
    __syncthreads();
    v16h Pf = load_ab_frag(&Plds[0], 32, lane);

#pragma unroll
    for (int nc = 0; nc < 4; ++nc) {
      v16h Vrf = load_b_colfrag(vr + (size_t)kbase * Dm + coff + nc * 16, Dm, lane);
      v16h Vif = load_b_colfrag(vi + (size_t)kbase * Dm + coff + nc * 16, Dm, lane);
      oR[nc] = WMMA_F16(Pf, Vrf, oR[nc]);
      oI[nc] = WMMA_F16(Pf, Vif, oI[nc]);
    }
    __syncthreads();
  }

  float invl[8];
#pragma unroll
  for (int r = 0; r < 8; ++r) invl[r] = 1.0f / lrow[r];
#pragma unroll
  for (int nc = 0; nc < 4; ++nc)
#pragma unroll
    for (int r = 0; r < 8; ++r) {
      int row = qbase + r + 8 * halfid;
      int col = coff + nc * 16 + lane15;
      size_t o = (size_t)row * Dm + col;
      outr[o] = (half_t)(oR[nc][r] * invl[r]);
      outi[o] = (half_t)(oI[nc][r] * invl[r]);
    }
}

// ---------------------------------------------------------------------------
extern "C" void kernel_launch(void* const* d_in, const int* in_sizes, int n_in,
                              void* d_out, int out_size, void* d_ws, size_t ws_size,
                              hipStream_t stream) {
  (void)in_sizes; (void)n_in; (void)out_size; (void)ws_size;
  const float* x_ri = (const float*)d_in[0];
  const float* g1  = (const float*)d_in[1];
  const float* b1  = (const float*)d_in[2];
  const float* wq  = (const float*)d_in[3];
  const float* bq  = (const float*)d_in[4];
  const float* wk  = (const float*)d_in[5];
  const float* bk  = (const float*)d_in[6];
  const float* wv  = (const float*)d_in[7];
  const float* bv  = (const float*)d_in[8];
  const float* wo  = (const float*)d_in[9];
  const float* bo  = (const float*)d_in[10];
  const float* g2  = (const float*)d_in[11];
  const float* b2  = (const float*)d_in[12];
  const float* w1  = (const float*)d_in[13];
  const float* bw1 = (const float*)d_in[14];
  const float* g3  = (const float*)d_in[15];
  const float* b3  = (const float*)d_in[16];
  const float* w2  = (const float*)d_in[17];
  const float* bw2 = (const float*)d_in[18];

  char* ws = (char*)d_ws;
  size_t off = 0;
  auto alloc = [&](size_t bytes) -> void* {
    void* p = ws + off;
    off = (off + bytes + 255) & ~(size_t)255;
    return p;
  };

  float*  xr0 = (float*) alloc((size_t)M_ * D_ * 4);
  float*  xi0 = (float*) alloc((size_t)M_ * D_ * 4);
  float*  xr1 = (float*) alloc((size_t)M_ * D_ * 4);
  float*  xi1 = (float*) alloc((size_t)M_ * D_ * 4);
  half_t* lnr = (half_t*)alloc((size_t)M_ * HID_ * 2);
  half_t* lni = (half_t*)alloc((size_t)M_ * HID_ * 2);
  half_t* qpr = (half_t*)alloc((size_t)M_ * D_ * 2);
  half_t* qpi = (half_t*)alloc((size_t)M_ * D_ * 2);
  half_t* kpr = (half_t*)alloc((size_t)M_ * D_ * 2);
  half_t* kpi = (half_t*)alloc((size_t)M_ * D_ * 2);
  half_t* vpr = (half_t*)alloc((size_t)M_ * D_ * 2);
  half_t* vpi = (half_t*)alloc((size_t)M_ * D_ * 2);
  half_t* aor = (half_t*)alloc((size_t)M_ * D_ * 2);
  half_t* aoi = (half_t*)alloc((size_t)M_ * D_ * 2);
  float*  h1r = (float*) alloc((size_t)M_ * HID_ * 4);
  float*  h1i = (float*) alloc((size_t)M_ * HID_ * 4);
  half_t* wqr = (half_t*)alloc((size_t)D_ * D_ * 2);
  half_t* wqi = (half_t*)alloc((size_t)D_ * D_ * 2);
  half_t* wkr = (half_t*)alloc((size_t)D_ * D_ * 2);
  half_t* wki = (half_t*)alloc((size_t)D_ * D_ * 2);
  half_t* wvr = (half_t*)alloc((size_t)D_ * D_ * 2);
  half_t* wvi = (half_t*)alloc((size_t)D_ * D_ * 2);
  half_t* wor = (half_t*)alloc((size_t)D_ * D_ * 2);
  half_t* woi = (half_t*)alloc((size_t)D_ * D_ * 2);
  half_t* w1r = (half_t*)alloc((size_t)HID_ * D_ * 2);
  half_t* w1i = (half_t*)alloc((size_t)HID_ * D_ * 2);
  half_t* w2r = (half_t*)alloc((size_t)D_ * HID_ * 2);
  half_t* w2i = (half_t*)alloc((size_t)D_ * HID_ * 2);

  const int nMD = M_ * D_;
  const int nDD = D_ * D_;
  const int nHD = HID_ * D_;

  // 1. deinterleave x -> residual planes
  deint_kernel<<<nMD / 256, 256, 0, stream>>>(x_ri, xr0, xi0, nMD);

  // 2. weights f32 interleaved -> f16 planar
  wconv_kernel<<<nDD / 256, 256, 0, stream>>>(wq, wqr, wqi, nDD);
  wconv_kernel<<<nDD / 256, 256, 0, stream>>>(wk, wkr, wki, nDD);
  wconv_kernel<<<nDD / 256, 256, 0, stream>>>(wv, wvr, wvi, nDD);
  wconv_kernel<<<nDD / 256, 256, 0, stream>>>(wo, wor, woi, nDD);
  wconv_kernel<<<nHD / 256, 256, 0, stream>>>(w1, w1r, w1i, nHD);
  wconv_kernel<<<nHD / 256, 256, 0, stream>>>(w2, w2r, w2i, nHD);

  // 3. LN1
  cln_kernel<<<M_ / 8, 256, 0, stream>>>(xr0, xi0, g1, b1, lnr, lni, D_);

  // 4. QKV projections (mode 0 -> f16 planar)
  dim3 gQ(D_ / 64, M_ / 128);
  cgemm_kernel<<<gQ, 256, 0, stream>>>(lnr, lni, wqr, wqi, bq, M_, D_, D_, 0,
                                       nullptr, nullptr, qpr, qpi, nullptr, nullptr, nullptr);
  cgemm_kernel<<<gQ, 256, 0, stream>>>(lnr, lni, wkr, wki, bk, M_, D_, D_, 0,
                                       nullptr, nullptr, kpr, kpi, nullptr, nullptr, nullptr);
  cgemm_kernel<<<gQ, 256, 0, stream>>>(lnr, lni, wvr, wvi, bv, M_, D_, D_, 0,
                                       nullptr, nullptr, vpr, vpi, nullptr, nullptr, nullptr);

  // 5. attention
  dim3 gA(B_ * H_, S_ / 16);
  cattn_kernel<<<gA, 32, 0, stream>>>(qpr, qpi, kpr, kpi, vpr, vpi, aor, aoi);

  // 6. O projection + residual (mode 1 -> f32 planar xr1/xi1)
  cgemm_kernel<<<gQ, 256, 0, stream>>>(aor, aoi, wor, woi, bo, M_, D_, D_, 1,
                                       xr0, xi0, nullptr, nullptr, xr1, xi1, nullptr);

  // 7. LN2
  cln_kernel<<<M_ / 8, 256, 0, stream>>>(xr1, xi1, g2, b2, lnr, lni, D_);

  // 8. FFN1 + CGELU (mode 2 -> f32 planar h1)
  dim3 gF1(HID_ / 64, M_ / 128);
  cgemm_kernel<<<gF1, 256, 0, stream>>>(lnr, lni, w1r, w1i, bw1, M_, HID_, D_, 2,
                                        nullptr, nullptr, nullptr, nullptr, h1r, h1i, nullptr);

  // 9. LN3 over hidden
  cln_kernel<<<M_ / 8, 256, 0, stream>>>(h1r, h1i, g3, b3, lnr, lni, HID_);

  // 10. FFN2 + CGELU + residual -> interleaved d_out
  cgemm_kernel<<<gQ, 256, 0, stream>>>(lnr, lni, w2r, w2i, bw2, M_, D_, HID_, 3,
                                       xr1, xi1, nullptr, nullptr, nullptr, nullptr,
                                       (float*)d_out);
}
